// TransformerModule_83150566851019
// MI455X (gfx1250) — matrix operand order repaired and checked by reading, not compiled, not run on hardware
//
#include <hip/hip_runtime.h>
#include <hip/hip_bf16.h>
#include <math.h>

// ---------------------------------------------------------------------------
// Equivariant graph transformer score model for MI455X (gfx1250, wave32).
// Heavy per-edge MLPs run on v_wmma_f32_16x16x32_f16 (fp16 in, fp32 acc).
// ~0.38 TFLOP total vs ~50 MB of traffic -> compute bound. Each layer is fully
// fused per destination node: operands staged in LDS (ds_load_b128 A-frags),
// weights pre-shuffled into per-lane WMMA fragment layout so each B-frag is
// one contiguous 32B load (2x global_load_b128, coalesced across the wave).
// ---------------------------------------------------------------------------

#define Bd   8
#define Nd_  1024
#define Kn   32      // kNN neighbors
#define Dm   256     // node width (= H*DH)
#define Hh   8
#define DH   32
#define TDIM 16
#define NB   32
#define HID  128
#define NS   128
#define EINP 320     // EIN=304 padded to 32
#define WVKP 288     // D+SHD=265 padded to 32

typedef _Float16 h16;
typedef __attribute__((ext_vector_type(16))) _Float16 v16h;
typedef __attribute__((ext_vector_type(8)))  float    v8f;

__device__ __forceinline__ v8f wmma16(v16h a, v16h b, v8f c) {
  // 8 args: (neg_a, A, neg_b, B, c_mod, C, reuse_a, reuse_b)
  return __builtin_amdgcn_wmma_f32_16x16x32_f16(false, a, false, b, (short)0, c,
                                                false, false);
}

__device__ __forceinline__ float gelu_f(float x) {
  const float c = 0.7978845608028654f;           // sqrt(2/pi)
  return 0.5f * x * (1.f + tanhf(c * (x + 0.044715f * x * x * x)));
}

// ---- fragment loaders (layouts per CDNA5 ISA 7.12.2) ----------------------
// A (16x32 f16) from row-major LDS: lane L<16 holds row M=L,
// K={k0..k0+7, k0+16..k0+23}; lanes>=16 same rows, pattern shifted +8.
// -> two contiguous 16B chunks per lane = 2x ds_load_b128.
__device__ __forceinline__ v16h load_a(const h16* base, int stride, int mrow,
                                       int kbase, int lane) {
  int m  = mrow + (lane & 15);
  int k0 = kbase + ((lane & 16) ? 8 : 0);
  const h16* p = base + m * stride + k0;
  v16h a;
#pragma unroll
  for (int j = 0; j < 8; ++j) { a[j] = p[j]; a[8 + j] = p[16 + j]; }
  return a;
}

// B fragment from pre-shuffled weights: [tile][lane][16 halves] contiguous.
__device__ __forceinline__ v16h load_bf(const h16* __restrict__ Wfrag,
                                        int tile, int lane) {
  return *(const v16h*)(Wfrag + (((size_t)tile * 32 + lane) << 4));
}

// ---------------------------------------------------------------------------
// pack f32 weights [l][Ksrc][ncols] into per-lane WMMA B fragments:
// dst[l][kc][nt][lane][j] = W[kc*32 + ((lane&16)?8:0) + (j&7) + ((j&8)?16:0)][nt*16 + (lane&15)]
__global__ void frag_pack_kernel(const float* __restrict__ src, h16* __restrict__ dst,
                                 int Ksrc, int KC, int NT, int ncols, int layers) {
  long i = (long)blockIdx.x * blockDim.x + threadIdx.x;
  long tot = (long)layers * KC * NT * 32 * 16;
  if (i >= tot) return;
  int  j    = (int)(i & 15);
  int  lane = (int)((i >> 4) & 31);
  long r    = i >> 9;
  int  nt   = (int)(r % NT); r /= NT;
  int  kc   = (int)(r % KC);
  int  l    = (int)(r / KC);
  int  c    = nt * 16 + (lane & 15);
  int  k    = kc * 32 + ((lane & 16) ? 8 : 0) + (j & 7) + ((j & 8) ? 16 : 0);
  float v = (k < Ksrc) ? src[((long)l * Ksrc + k) * ncols + c] : 0.f;
  dst[i] = (h16)v;
}

// ---------------------------------------------------------------------------
// kNN: one block per (b,n); 1024 d^2 in LDS, 32 argmin rounds (ties -> low idx)
__global__ __launch_bounds__(128) void knn_kernel(const float* __restrict__ x,
                                                  int* __restrict__ idx) {
  int b = blockIdx.x >> 10, n = blockIdx.x & 1023, tid = threadIdx.x;
  __shared__ float d2[Nd_];
  __shared__ float mv[128];
  __shared__ int   mi[128];
  __shared__ float px[3];
  if (tid < 3) px[tid] = x[((size_t)b * Nd_ + n) * 3 + tid];
  __syncthreads();
  float p0 = px[0], p1 = px[1], p2 = px[2];
  for (int j = tid; j < Nd_; j += 128) {
    const float* q = x + ((size_t)b * Nd_ + j) * 3;
    float dx = p0 - q[0], dy = p1 - q[1], dz = p2 - q[2];
    d2[j] = dx * dx + dy * dy + dz * dz;
  }
  __syncthreads();
  for (int r = 0; r < Kn; ++r) {
    float bv = 1e30f; int bi = 1 << 20;
    for (int j = tid; j < Nd_; j += 128) {
      float v = d2[j];
      if (v < bv || (v == bv && j < bi)) { bv = v; bi = j; }
    }
    mv[tid] = bv; mi[tid] = bi;
    __syncthreads();
    for (int s = 64; s > 0; s >>= 1) {
      if (tid < s) {
        float ov = mv[tid + s]; int oi = mi[tid + s];
        if (ov < mv[tid] || (ov == mv[tid] && oi < mi[tid])) { mv[tid] = ov; mi[tid] = oi; }
      }
      __syncthreads();
    }
    if (tid == 0) { idx[(size_t)blockIdx.x * Kn + r] = mi[0]; d2[mi[0]] = 1e30f; }
    __syncthreads();
  }
}

// ---------------------------------------------------------------------------
// per-edge geometric features: edge_base[48] (rbf*cut + t) fp16, sh[9] fp16, cut
__global__ void edge_kernel(const float* __restrict__ x, const float* __restrict__ t,
                            const int* __restrict__ idx, float* __restrict__ cut,
                            h16* __restrict__ eb16, h16* __restrict__ sh16) {
  int e = blockIdx.x * blockDim.x + threadIdx.x;
  if (e >= Bd * Nd_ * Kn) return;
  int bn = e >> 5, b = bn >> 10, n = bn & 1023;
  int s = idx[e];
  const float* pd = x + ((size_t)b * Nd_ + n) * 3;
  const float* ps = x + ((size_t)b * Nd_ + s) * 3;
  float vx = pd[0] - ps[0], vy = pd[1] - ps[1], vz = pd[2] - ps[2];
  float r  = sqrtf(vx * vx + vy * vy + vz * vz);
  float rr = fmaxf(r, 1e-9f);
  float ux = vx / rr, uy = vy / rr, uz = vz / rr;
  const float s3 = 1.7320508075688772f, s15 = 3.872983346207417f, s5 = 2.23606797749979f;
  float sh[9];
  sh[0] = 1.f;
  sh[1] = s3 * ux; sh[2] = s3 * uy; sh[3] = s3 * uz;
  sh[4] = s15 * ux * uy; sh[5] = s15 * uy * uz;
  sh[6] = 0.5f * s5 * (3.f * uz * uz - 1.f);
  sh[7] = s15 * ux * uz; sh[8] = 0.5f * s15 * (ux * ux - uy * uy);
  float xs = 10.f * (1.f - 0.5f * r);
  float c  = (xs > 0.f) ? 1.4f * __expf(-1.f / xs) : 0.f;
  cut[e] = c;
  const float step = 2.0f / 31.0f;
  const float amp  = 5.656854249492381f * 0.95f / 1.12f;  // sqrt(NB)*0.95/1.12
  h16* eb = eb16 + (size_t)e * 48;
  for (int i = 0; i < NB; ++i) {
    float d = (r - step * i) / step;
    eb[i] = (h16)(amp * __expf(-d * d) * c);
  }
  const float* tb = t + (size_t)b * TDIM;
  for (int i = 0; i < TDIM; ++i) eb[NB + i] = (h16)tb[i];
  h16* shp = sh16 + (size_t)e * 16;
  for (int i = 0; i < 9; ++i)  shp[i] = (h16)sh[i];
  for (int i = 9; i < 16; ++i) shp[i] = (h16)0.f;
}

// ---------------------------------------------------------------------------
// node embedding: [y, t] @ W_embed  (19 MACs/output -> VALU)
__global__ void embed_kernel(const float* __restrict__ y, const float* __restrict__ t,
                             const float* __restrict__ W, float* __restrict__ node) {
  int i = blockIdx.x * blockDim.x + threadIdx.x;
  if (i >= Bd * Nd_ * Dm) return;
  int o = i & 255, bn = i >> 8, b = bn >> 10;
  const float* f  = y + (size_t)bn * 3;
  const float* tb = t + (size_t)b * TDIM;
  float s = f[0] * W[0 * Dm + o] + f[1] * W[1 * Dm + o] + f[2] * W[2 * Dm + o];
  for (int j = 0; j < TDIM; ++j) s += tb[j] * W[(3 + j) * Dm + o];
  node[i] = s;
}

// ---------------------------------------------------------------------------
// fully fused transformer layer: one workgroup (4 wave32) per destination node
__global__ __launch_bounds__(128)
void layer_kernel(const float* __restrict__ node_in, float* __restrict__ node_out,
                  const int* __restrict__ idx, const float* __restrict__ cutg,
                  const h16* __restrict__ eb16, const h16* __restrict__ sh16,
                  const h16* __restrict__ wk1f, const float* __restrict__ bk1,
                  const h16* __restrict__ wk2f, const float* __restrict__ wq,
                  const h16* __restrict__ wvf, const float* __restrict__ wo,
                  int last) {
  int bn = blockIdx.x, tid = threadIdx.x;
  int wave = tid >> 5, lane = tid & 31;
  int bbase = (bn >> 10) << 10;  // b*N

  // LDS pool: Ebuf [32x320] is dead after stage 1 -> KV [32x256] aliases it.
  __shared__ h16   POOL0[Kn * EINP];   // Ebuf, then KV
  __shared__ h16   VB[Kn * WVKP];      // [node[src](256), sh(9), pad] per edge
  __shared__ h16   H1[Kn * HID];       // gelu(e@Wk1+b) fp16
  __shared__ float selfF[256];
  __shared__ float qv[256];
  __shared__ float cutS[Kn];
  __shared__ float logit[Kn * Hh];
  __shared__ float alpha[Kn * Hh];
  __shared__ float agg[256];
  __shared__ int   srcIdx[Kn];
  h16* Ebuf = POOL0;
  h16* KV   = POOL0;

  if (tid < Kn) {
    srcIdx[tid] = idx[(size_t)bn * Kn + tid];
    cutS[tid]   = cutg[(size_t)bn * Kn + tid];
  }
  for (int i = tid; i < 256; i += 128) selfF[i] = node_in[(size_t)bn * Dm + i];
  __syncthreads();

  // ---- build E [32 x 320] = [edge_base(48) | node[src][:128] | node[self][:128] | 0]
  for (int i = tid; i < Kn * EINP; i += 128) {
    int k = i / EINP, c = i - k * EINP;
    h16 v;
    if (c < 48)       v = eb16[(size_t)bn * Kn * 48 + k * 48 + c];
    else if (c < 176) v = (h16)node_in[((size_t)(bbase + srcIdx[k])) * Dm + (c - 48)];
    else if (c < 304) v = (h16)selfF[c - 176];
    else              v = (h16)0.f;
    Ebuf[i] = v;
  }
  // ---- build V-input [32 x 288] = [node[src](256) | sh(9) | 0]
  for (int i = tid; i < Kn * WVKP; i += 128) {
    int k = i / WVKP, c = i - k * WVKP;
    h16 v;
    if (c < 256)      v = (h16)node_in[((size_t)(bbase + srcIdx[k])) * Dm + c];
    else if (c < 265) v = sh16[(size_t)bn * Kn * 16 + k * 16 + (c - 256)];
    else              v = (h16)0.f;
    VB[i] = v;
  }
  // q = node_self @ Wq (1-row matvec, coalesced across threads; ~1% of FLOPs)
  for (int o = tid; o < 256; o += 128) {
    float s = 0.f;
    for (int d = 0; d < 256; ++d) s += selfF[d] * wq[(size_t)d * 256 + o];
    qv[o] = s;
  }
  __syncthreads();

  // ---- stage 1: H1 = gelu(E @ Wk1 + b): 16 tiles x 10 k-chunks
  for (int t = wave; t < 16; t += 4) {
    int mt = t >> 3, nt = t & 7;
    v8f acc = {};
#pragma unroll
    for (int kc = 0; kc < EINP / 32; ++kc) {
      v16h a = load_a(Ebuf, EINP, mt * 16, kc * 32, lane);
      v16h b = load_bf(wk1f, kc * 8 + nt, lane);
      acc = wmma16(a, b, acc);
    }
    int nn = nt * 16 + (lane & 15);
    int mb = mt * 16 + ((lane & 16) ? 8 : 0);
    float bias = bk1[nn];
#pragma unroll
    for (int v = 0; v < 8; ++v)
      H1[(mb + v) * HID + nn] = (h16)gelu_f(acc[v] + bias);
  }
  __syncthreads();  // Ebuf dead from here; KV may overwrite POOL0

  // ---- stage 2: KF = H1 @ Wk2  [32 x 256]: 32 tiles x 4 k-chunks
  for (int t = wave; t < 32; t += 4) {
    int mt = t >> 4, nt = t & 15;
    v8f acc = {};
#pragma unroll
    for (int kc = 0; kc < HID / 32; ++kc) {
      v16h a = load_a(H1, HID, mt * 16, kc * 32, lane);
      v16h b = load_bf(wk2f, kc * 16 + nt, lane);
      acc = wmma16(a, b, acc);
    }
    int nn = nt * 16 + (lane & 15);
    int mb = mt * 16 + ((lane & 16) ? 8 : 0);
#pragma unroll
    for (int v = 0; v < 8; ++v) KV[(mb + v) * 256 + nn] = (h16)acc[v];
  }
  __syncthreads();

  // ---- logits + cutoff-weighted softmax over K per head
  for (int li = tid; li < Kn * Hh; li += 128) {
    int k = li >> 3, h = li & 7;
    float s = 0.f;
    for (int d = 0; d < DH; ++d) s += qv[h * DH + d] * (float)KV[k * 256 + h * DH + d];
    logit[li] = s * 0.17677669529663687f;  // 1/sqrt(DH)
  }
  __syncthreads();
  if (tid < Hh) {
    int h = tid;
    float m = -1e30f;
    for (int k = 0; k < Kn; ++k) m = fmaxf(m, logit[k * 8 + h]);
    float sum = 0.f;
    for (int k = 0; k < Kn; ++k) {
      float w = cutS[k] * __expf(logit[k * 8 + h] - m);
      alpha[k * 8 + h] = w; sum += w;
    }
    float inv = 1.f / (sum + 1e-9f);
    for (int k = 0; k < Kn; ++k) alpha[k * 8 + h] *= inv;
  }
  __syncthreads();

  // ---- stage 3: V = VB @ Wv  [32 x 256]: 32 tiles x 9 k-chunks (reuse KV)
  for (int t = wave; t < 32; t += 4) {
    int mt = t >> 4, nt = t & 15;
    v8f acc = {};
#pragma unroll
    for (int kc = 0; kc < WVKP / 32; ++kc) {
      v16h a = load_a(VB, WVKP, mt * 16, kc * 32, lane);
      v16h b = load_bf(wvf, kc * 16 + nt, lane);
      acc = wmma16(a, b, acc);
    }
    int nn = nt * 16 + (lane & 15);
    int mb = mt * 16 + ((lane & 16) ? 8 : 0);
#pragma unroll
    for (int v = 0; v < 8; ++v) KV[(mb + v) * 256 + nn] = (h16)acc[v];
  }
  __syncthreads();

  // ---- aggregate: agg[c] = sum_k alpha[k,h]*V[k,c]
  for (int o = tid; o < 256; o += 128) {
    int h = o >> 5;
    float s = 0.f;
    for (int k = 0; k < Kn; ++k) s += alpha[k * 8 + h] * (float)KV[k * 256 + o];
    agg[o] = s;
  }
  __syncthreads();

  // ---- output projection
  if (!last) {
    for (int o = tid; o < 256; o += 128) {
      float s = selfF[o];
      for (int c = 0; c < 256; ++c) s += agg[c] * wo[(size_t)c * 256 + o];
      float r = (o < 64) ? gelu_f(s) : (o < NS ? tanhf(s) : s);
      node_out[(size_t)bn * Dm + o] = r;
    }
  } else {
    if (tid < 3) {
      float s = 0.f;
      for (int c = 0; c < 256; ++c) s += agg[c] * wo[c * 3 + tid];
      node_out[(size_t)bn * 3 + tid] = s;
    }
  }
}

// ---------------------------------------------------------------------------
extern "C" void kernel_launch(void* const* d_in, const int* in_sizes, int n_in,
                              void* d_out, int out_size, void* d_ws, size_t ws_size,
                              hipStream_t stream) {
  const float* x       = (const float*)d_in[0];
  const float* y       = (const float*)d_in[1];
  const float* t       = (const float*)d_in[2];
  const float* W_embed = (const float*)d_in[3];
  const float* Wk1     = (const float*)d_in[4];
  const float* bk1     = (const float*)d_in[5];
  const float* Wk2     = (const float*)d_in[6];
  const float* Wq      = (const float*)d_in[7];
  const float* Wv      = (const float*)d_in[8];
  const float* Wo      = (const float*)d_in[9];
  const float* Wo_out  = (const float*)d_in[10];
  float* out = (float*)d_out;

  // workspace carve-out (~53 MB total)
  char* ws = (char*)d_ws;
  size_t off = 0;
  auto take = [&](size_t bytes) -> char* {
    char* p = ws + off;
    off = (off + bytes + 255) & ~(size_t)255;
    return p;
  };
  int*   idx   = (int*)  take((size_t)Bd * Nd_ * Kn * 4);
  float* cut   = (float*)take((size_t)Bd * Nd_ * Kn * 4);
  h16*   eb16  = (h16*)  take((size_t)Bd * Nd_ * Kn * 48 * 2);
  h16*   sh16  = (h16*)  take((size_t)Bd * Nd_ * Kn * 16 * 2);
  float* nodeA = (float*)take((size_t)Bd * Nd_ * Dm * 4);
  float* nodeB = (float*)take((size_t)Bd * Nd_ * Dm * 4);
  h16*   wk1f  = (h16*)  take((size_t)4 * (EINP / 32) * (HID / 16) * 512 * 2);
  h16*   wk2f  = (h16*)  take((size_t)4 * (HID / 32) * (256 / 16) * 512 * 2);
  h16*   wvf   = (h16*)  take((size_t)4 * (WVKP / 32) * (256 / 16) * 512 * 2);

  // weight fp16 convert + K-pad + shuffle into per-lane WMMA B fragments
  {
    long n1 = 4L * (EINP / 32) * (HID / 16) * 512;
    long n2 = 4L * (HID / 32) * (256 / 16) * 512;
    long n3 = 4L * (WVKP / 32) * (256 / 16) * 512;
    frag_pack_kernel<<<(int)((n1 + 255) / 256), 256, 0, stream>>>(Wk1, wk1f, 304, EINP / 32, HID / 16, HID, 4);
    frag_pack_kernel<<<(int)((n2 + 255) / 256), 256, 0, stream>>>(Wk2, wk2f, HID, HID / 32, 256 / 16, 256, 4);
    frag_pack_kernel<<<(int)((n3 + 255) / 256), 256, 0, stream>>>(Wv, wvf, 265, WVKP / 32, 256 / 16, 256, 4);
  }

  knn_kernel<<<Bd * Nd_, 128, 0, stream>>>(x, idx);
  edge_kernel<<<(Bd * Nd_ * Kn + 255) / 256, 256, 0, stream>>>(x, t, idx, cut, eb16, sh16);
  embed_kernel<<<(Bd * Nd_ * Dm + 255) / 256, 256, 0, stream>>>(y, t, W_embed, nodeA);

  const size_t w1l = (size_t)(EINP / 32) * (HID / 16) * 512;
  const size_t w2l = (size_t)(HID / 32) * (256 / 16) * 512;
  const size_t w3l = (size_t)(WVKP / 32) * (256 / 16) * 512;

  float* nin = nodeA;
  float* nout = nodeB;
  for (int l = 0; l < 3; ++l) {
    layer_kernel<<<Bd * Nd_, 128, 0, stream>>>(
        nin, nout, idx, cut, eb16, sh16,
        wk1f + (size_t)l * w1l, bk1 + l * HID,
        wk2f + (size_t)l * w2l, Wq + (size_t)l * 256 * 256,
        wvf + (size_t)l * w3l, Wo + (size_t)l * 256 * 256, 0);
    float* tmp = nin; nin = nout; nout = tmp;
  }
  layer_kernel<<<Bd * Nd_, 128, 0, stream>>>(
      nin, out, idx, cut, eb16, sh16,
      wk1f + (size_t)3 * w1l, bk1 + 3 * HID,
      wk2f + (size_t)3 * w2l, Wq + (size_t)3 * 256 * 256,
      wvf + (size_t)3 * w3l, Wo_out, 1);
}
